// VectorQuantizer_33775622816341
// MI455X (gfx1250) — matrix operand order repaired and checked by reading, not compiled, not run on hardware
//
#include <hip/hip_runtime.h>

#define N_DIM 32768
#define D_DIM 512
#define K_DIM 4096
#define M_TILE 64    // rows of x per workgroup (4 waves x 16 rows)
#define WAVES  4
#define KT_PER 4     // K-tiles (16 cols each) processed concurrently per wave

typedef __attribute__((ext_vector_type(16))) __bf16 v16bf;
typedef __attribute__((ext_vector_type(8)))  __bf16 v8bf;
typedef __attribute__((ext_vector_type(8)))  float  v8f;

union ABFrag { v16bf v; struct { v8bf lo, hi; } p; };

// ---------------------------------------------------------------------------
// Kernel 1: transpose embeddings [D][K] -> eT [K][D] (f32) and split into
// bf16 hi/lo halves for the 3-pass split-bf16 WMMA GEMM.
// t indexes [K][D] with d fastest -> coalesced writes; reads hit L2 (8 MB).
// ---------------------------------------------------------------------------
__global__ void vq_convert_embed(const float* __restrict__ emb,
                                 __bf16* __restrict__ ehi,
                                 __bf16* __restrict__ elo,
                                 float* __restrict__ eT) {
  int t = blockIdx.x * blockDim.x + threadIdx.x;   // exact grid: K*D threads
  int k = t >> 9;                                  // t / D_DIM
  int d = t & (D_DIM - 1);
  float v = emb[(size_t)d * K_DIM + k];
  __bf16 h = (__bf16)v;
  ehi[t] = h;
  elo[t] = (__bf16)(v - (float)h);
  eT[t]  = v;
}

// ---------------------------------------------------------------------------
// Kernel 2: ||e_k||^2 per codebook column, one wave per column.
// ---------------------------------------------------------------------------
__global__ void vq_enorm(const float* __restrict__ eT, float* __restrict__ en) {
  int k = blockIdx.x * (blockDim.x >> 5) + (threadIdx.x >> 5);
  int lane = threadIdx.x & 31;
  float s = 0.0f;
  for (int d = lane; d < D_DIM; d += 32) {
    float v = eT[(size_t)k * D_DIM + d];
    s += v * v;
  }
  #pragma unroll
  for (int off = 16; off > 0; off >>= 1) s += __shfl_xor(s, off, 32);
  if (lane == 0) en[k] = s;
}

__global__ void vq_zero_loss(float* loss) {
  if (threadIdx.x == 0 && blockIdx.x == 0) *loss = 0.0f;
}

// ---------------------------------------------------------------------------
// Kernel 3: fused split-bf16 WMMA GEMM + argmin over K.
// WG: 128 threads = 4 waves; wave w owns 16 rows (B stream shared by all 4
// waves via WGP$). x tile staged once in LDS as bf16 hi/lo (128 KB).
// Each wave sweeps K in groups of 4 x 16 columns: one A hi/lo fragment pair
// feeds 4 independent accumulator chains (12 WMMAs per A load), giving the
// XDL pipe 4 independent v_wmma_f32_16x16x32_bf16 streams to interleave.
// Distances dist = ||e||^2 - 2 x.e tracked as per-lane running (min, idx),
// reduced across the 16 lanes of each C-matrix half at the end.
// ---------------------------------------------------------------------------
__global__ __launch_bounds__(32 * WAVES) void vq_gemm_argmin(
    const float* __restrict__ x,
    const __bf16* __restrict__ ehi,   // [K][D]
    const __bf16* __restrict__ elo,   // [K][D]
    const float* __restrict__ enorm,  // [K]
    int* __restrict__ idxOut) {       // [N]
  __shared__ __align__(32) __bf16 shi[M_TILE * D_DIM];  // 64 KB
  __shared__ __align__(32) __bf16 slo[M_TILE * D_DIM];  // 64 KB

  // Stage & split x tile: 64 rows x 512, coalesced.
  for (int t = threadIdx.x; t < M_TILE * D_DIM; t += 32 * WAVES) {
    int row = t >> 9;
    int d = t & (D_DIM - 1);
    float v = x[((size_t)blockIdx.x * M_TILE + row) * D_DIM + d];
    __bf16 h = (__bf16)v;
    shi[t] = h;
    slo[t] = (__bf16)(v - (float)h);
  }
  __syncthreads();

  const int lane = threadIdx.x & 31;
  const int wv   = threadIdx.x >> 5;
  const int lr   = wv * 16 + (lane & 15);  // A-matrix row for this lane
  const int sel  = lane >> 4;              // lane half (K-range selector)

  float mv[8];
  int   mi[8];
  #pragma unroll
  for (int r = 0; r < 8; ++r) { mv[r] = 3.4e38f; mi[r] = 0; }

  for (int kt = 0; kt < K_DIM / (16 * KT_PER); ++kt) {
    const int k0 = kt * 16 * KT_PER;
    const int colBase = k0 + (lane & 15);  // B column for j=0
    v8f acc[KT_PER];
    #pragma unroll
    for (int j = 0; j < KT_PER; ++j) acc[j] = (v8f){};

    const __bf16* bhp = ehi + (size_t)colBase * D_DIM + sel * 16;
    const __bf16* blp = elo + (size_t)colBase * D_DIM + sel * 16;

    #pragma unroll 2
    for (int d0 = 0; d0 < D_DIM; d0 += 32) {
      // A fragment 16x32 bf16: lanes<16 hold K=d0+0..15,
      // lanes>=16 hold K=d0+16..31 (contiguous, matches B).
      ABFrag ah, al;
      const __bf16* ap  = &shi[lr * D_DIM + d0 + sel * 16];
      ah.p.lo = *(const v8bf*)(ap);
      ah.p.hi = *(const v8bf*)(ap + 8);
      const __bf16* alp = &slo[lr * D_DIM + d0 + sel * 16];
      al.p.lo = *(const v8bf*)(alp);
      al.p.hi = *(const v8bf*)(alp + 8);

      #pragma unroll
      for (int j = 0; j < KT_PER; ++j) {
        // B fragment 32x16 bf16: lanes<16 hold K=d0+0..15, lanes>=16
        // d0+16..31; contiguous in the eT row -> one 32-byte load each.
        // Column j offset folds into the 24-bit instruction offset.
        v16bf bh = *(const v16bf*)(bhp + (size_t)j * 16 * D_DIM + d0);
        v16bf bl = *(const v16bf*)(blp + (size_t)j * 16 * D_DIM + d0);
        acc[j] = __builtin_amdgcn_wmma_f32_16x16x32_bf16(
            false, ah.v, false, bh, (short)0, acc[j], false, false);
        acc[j] = __builtin_amdgcn_wmma_f32_16x16x32_bf16(
            false, ah.v, false, bl, (short)0, acc[j], false, false);
        acc[j] = __builtin_amdgcn_wmma_f32_16x16x32_bf16(
            false, al.v, false, bh, (short)0, acc[j], false, false);
      }
    }

    // dist = ||e||^2 - 2*x.e  (||x||^2 constant per row, irrelevant to argmin)
    // j ascending + strict '<' preserves first-occurrence argmin semantics.
    #pragma unroll
    for (int j = 0; j < KT_PER; ++j) {
      const int col = colBase + 16 * j;
      float en = enorm[col];
      #pragma unroll
      for (int r = 0; r < 8; ++r) {
        float dv = __builtin_fmaf(-2.0f, acc[j][r], en);
        if (dv < mv[r]) { mv[r] = dv; mi[r] = col; }
      }
    }
  }

  // Cross-lane argmin within each 16-lane half (halves hold different rows).
  #pragma unroll
  for (int r = 0; r < 8; ++r) {
    #pragma unroll
    for (int off = 1; off < 16; off <<= 1) {
      float ov = __shfl_xor(mv[r], off, 32);
      int   oi = __shfl_xor(mi[r], off, 32);
      if (ov < mv[r] || (ov == mv[r] && oi < mi[r])) { mv[r] = ov; mi[r] = oi; }
    }
  }
  if ((lane & 15) == 0) {
    // lanes 0-15 hold rows M=r, lanes 16-31 hold rows M=r+8 (C layout).
    int base = blockIdx.x * M_TILE + wv * 16 + sel * 8;
    #pragma unroll
    for (int r = 0; r < 8; ++r) idxOut[base + r] = mi[r];
  }
}

// ---------------------------------------------------------------------------
// Kernel 4: gather quantized rows (coalesced from eT) + fused loss
// loss = 1.25 * mean((q - x)^2)   [commitment(0.25) + codebook(1.0)]
// ---------------------------------------------------------------------------
__global__ void vq_gather_loss(const float* __restrict__ x,
                               const float* __restrict__ eT,   // [K][D]
                               const int* __restrict__ idx,    // [N]
                               float* __restrict__ outq,       // [N][D]
                               float* __restrict__ loss) {
  size_t t = (size_t)blockIdx.x * blockDim.x + threadIdx.x;  // exact N*D grid
  int n = (int)(t >> 9);
  int d = (int)(t & (D_DIM - 1));
  float q  = eT[(size_t)idx[n] * D_DIM + d];
  float xv = x[t];
  outq[t] = q;
  float diff = q - xv;
  float p = diff * diff;
  #pragma unroll
  for (int off = 16; off > 0; off >>= 1) p += __shfl_xor(p, off, 32);
  __shared__ float red[8];
  int lane = threadIdx.x & 31, w = threadIdx.x >> 5;
  if (lane == 0) red[w] = p;
  __syncthreads();
  if (w == 0) {
    float s = (lane < 8) ? red[lane] : 0.0f;
    #pragma unroll
    for (int off = 4; off > 0; off >>= 1) s += __shfl_xor(s, off, 32);
    if (lane == 0) atomicAdd(loss, s * (1.25f / 16777216.0f));  // 1.25/(N*D)
  }
}

extern "C" void kernel_launch(void* const* d_in, const int* in_sizes, int n_in,
                              void* d_out, int out_size, void* d_ws, size_t ws_size,
                              hipStream_t stream) {
  (void)in_sizes; (void)n_in; (void)out_size; (void)ws_size;
  const float* x   = (const float*)d_in[0];  // [N][D]
  const float* emb = (const float*)d_in[1];  // [D][K]
  float* outq = (float*)d_out;
  float* loss = outq + (size_t)N_DIM * D_DIM;

  // Workspace layout (~16.2 MB):
  char* ws = (char*)d_ws;
  __bf16* ehi  = (__bf16*)(ws);                                  // 4 MB
  __bf16* elo  = (__bf16*)(ws + (4u << 20));                     // 4 MB
  float*  eT   = (float*) (ws + (8u << 20));                     // 8 MB
  float*  enrm = (float*) (ws + (16u << 20));                    // 16 KB
  int*    idx  = (int*)   (ws + (16u << 20) + (64u << 10));      // 128 KB

  vq_convert_embed<<<(K_DIM * D_DIM) / 256, 256, 0, stream>>>(emb, ehi, elo, eT);
  vq_enorm<<<K_DIM / 8, 256, 0, stream>>>(eT, enrm);
  vq_zero_loss<<<1, 32, 0, stream>>>(loss);
  vq_gemm_argmin<<<N_DIM / M_TILE, 32 * WAVES, 0, stream>>>(x, ehi, elo, enrm, idx);
  vq_gather_loss<<<(N_DIM * D_DIM) / 256, 256, 0, stream>>>(x, eT, idx, outq, loss);
}
